// reinforce_cons_loss_51479478010383
// MI455X (gfx1250) — matrix operand order repaired and checked by reading, not compiled, not run on hardware
//
#include <hip/hip_runtime.h>
#include <math.h>

// Problem geometry (fixed by the reference): preds [4,4,512,512] f32.
#define HH 512
#define WW 512
#define NIMG 4
#define NCLSS 4
#define NSAMP 10
#define NPLANE (NIMG * NSAMP)     // 40 sample planes
#define HWP (HH * WW)
#define TILES_X (WW / 16)
#define TILES_Y (HH / 16)
#define NTILES (TILES_X * TILES_Y) // 1024 16x16 tiles per plane
#define FLOOD_OUTER 64

typedef __attribute__((ext_vector_type(16))) _Float16 v16h;
typedef __attribute__((ext_vector_type(8)))  _Float16 v8h;
typedef __attribute__((ext_vector_type(8)))  float    v8f;

// ---------------------------------------------------------------- RNG --------
__device__ __forceinline__ unsigned pcg_hash(unsigned x) {
    x = x * 747796405u + 2891336453u;
    unsigned w = ((x >> ((x >> 28u) + 4u)) ^ x) * 277803737u;
    return (w >> 22u) ^ w;
}

// ------------------------------------------------- WMMA box-sum (16x16 tile) -
// Band matrix element: 1 iff patch index k is within +/-RAD of output index j
// (patch carries an 8-halo).  Used as A-operand (column pass) and B-operand
// (row pass) of V_WMMA_F32_16X16X32_F16.
template <int RAD>
__device__ __forceinline__ _Float16 bandv(int k, int j) {
    return (k >= j + 8 - RAD && k <= j + 8 + RAD) ? (_Float16)1.f : (_Float16)0.f;
}

// patch: 32x32 f16 LDS tile (8-halo).  rs: 16x32 f16 LDS scratch, [n][k] layout
// so both the D->rs stores and the rs->B reloads are contiguous b128 DS ops.
// Returns the (2*RAD+1)^2 SAME box sum of the central 16x16 tile in C/D layout
// (lane n=m, half hf; VGPR v -> element [M = v+8*hf][N = m]).
template <int RAD>
__device__ __forceinline__ v8f boxsum_wmma(const _Float16* patch, _Float16* rs) {
    const int lane = threadIdx.x & 31;
    const int m  = lane & 15;
    const int hf = lane >> 4;
    v16h aband, bband;
    v8f zero = {};
#pragma unroll
    for (int v = 0; v < 8; ++v) {
        // A-operand K index for VGPR v (lane-half selects K 0-15 / 16-31, matches rs reload)
        const int kb = 16 * hf + 2 * v;
        // B-operand K index for VGPR v (table order, matches patch A loads)
        const int k0 = (v < 4 ? 2 * v : 16 + 2 * (v - 4)) + 8 * hf;
        aband[2 * v]     = bandv<RAD>(kb,     m);   // A2[i=m][k]
        aband[2 * v + 1] = bandv<RAD>(kb + 1, m);
        bband[2 * v]     = bandv<RAD>(k0,     m);   // B[k][n=m]
        bband[2 * v + 1] = bandv<RAD>(k0 + 1, m);
    }
    // A operands: per lane two contiguous 16B chunks of patch row m / m+16.
    const v8h a0lo = *(const v8h*)&patch[m * 32 + 8 * hf];
    const v8h a0hi = *(const v8h*)&patch[m * 32 + 16 + 8 * hf];
    const v8h a1lo = *(const v8h*)&patch[(16 + m) * 32 + 8 * hf];
    const v8h a1hi = *(const v8h*)&patch[(16 + m) * 32 + 16 + 8 * hf];
    const v16h a0 = __builtin_shufflevector(a0lo, a0hi,
                    0, 1, 2, 3, 4, 5, 6, 7, 8, 9, 10, 11, 12, 13, 14, 15);
    const v16h a1 = __builtin_shufflevector(a1lo, a1hi,
                    0, 1, 2, 3, 4, 5, 6, 7, 8, 9, 10, 11, 12, 13, 14, 15);
    // Row pass: rowsum[r][j] = sum_{|d|<=RAD} patch[r][j+8+d]  (two 16-row halves)
    v8f d0 = __builtin_amdgcn_wmma_f32_16x16x32_f16(false, a0, false, bband,
                                                    (short)0, zero, false, false);
    v8f d1 = __builtin_amdgcn_wmma_f32_16x16x32_f16(false, a1, false, bband,
                                                    (short)0, zero, false, false);
    __syncthreads();                 // protect rs reuse across calls
    // D layout -> rs[n][k]; each lane stores 8 contiguous halves (values exact f16)
    v8h h0, h1;
#pragma unroll
    for (int v = 0; v < 8; ++v) { h0[v] = (_Float16)d0[v]; h1[v] = (_Float16)d1[v]; }
    *(v8h*)&rs[m * 32 + 8 * hf]      = h0;   // k = 8*hf + v
    *(v8h*)&rs[m * 32 + 16 + 8 * hf] = h1;   // k = 16 + 8*hf + v
    __syncthreads();
    // Reload in B layout: 16 contiguous halves starting at K = 16*hf.
    const v16h bcol = *(const v16h*)&rs[m * 32 + 16 * hf];
    // Column pass: out[i][j] = sum_{|d|<=RAD} rowsum[i+8+d][j]
    return __builtin_amdgcn_wmma_f32_16x16x32_f16(false, aband, false, bcol,
                                                  (short)0, zero, false, false);
}

// Patch loader: row `lane` of a 32x32 byte neighborhood -> f16 0/1 mask.
// Fast path (interior rows, 8B-aligned): 4x uint2 loads instead of 32 byte loads.
__device__ __forceinline__ void load_mask_row(_Float16* prow,
                                              const unsigned char* base,
                                              int gr, int gc0, int cls) {
    if (gr >= 0 && gr < HH && gc0 >= 0 && gc0 + 32 <= WW) {
        const uint2* src = (const uint2*)(base + gr * WW + gc0);
#pragma unroll
        for (int q = 0; q < 4; ++q) {
            const uint2 w = src[q];
#pragma unroll
            for (int b = 0; b < 4; ++b) {
                prow[q * 8 + b] = (((w.x >> (8 * b)) & 0xFFu) == (unsigned)cls)
                                      ? (_Float16)1.f : (_Float16)0.f;
                prow[q * 8 + 4 + b] = (((w.y >> (8 * b)) & 0xFFu) == (unsigned)cls)
                                      ? (_Float16)1.f : (_Float16)0.f;
            }
        }
    } else {
        for (int c = 0; c < 32; ++c) {
            const int gc = gc0 + c;
            _Float16 v = (_Float16)0.f;
            if (gr >= 0 && gr < HH && gc >= 0 && gc < WW)
                v = (base[gr * WW + gc] == (unsigned char)cls) ? (_Float16)1.f
                                                               : (_Float16)0.f;
            prow[c] = v;
        }
    }
}

// Same, but nonzero-byte -> 1 (for the flood-fill reach plane).
__device__ __forceinline__ void load_bin_row(_Float16* prow,
                                             const unsigned char* base,
                                             int gr, int gc0) {
    if (gr >= 0 && gr < HH && gc0 >= 0 && gc0 + 32 <= WW) {
        const uint2* src = (const uint2*)(base + gr * WW + gc0);
#pragma unroll
        for (int q = 0; q < 4; ++q) {
            const uint2 w = src[q];
#pragma unroll
            for (int b = 0; b < 4; ++b) {
                prow[q * 8 + b] = ((w.x >> (8 * b)) & 0xFFu) ? (_Float16)1.f
                                                             : (_Float16)0.f;
                prow[q * 8 + 4 + b] = ((w.y >> (8 * b)) & 0xFFu) ? (_Float16)1.f
                                                                 : (_Float16)0.f;
            }
        }
    } else {
        for (int c = 0; c < 32; ++c) {
            const int gc = gc0 + c;
            _Float16 v = (_Float16)0.f;
            if (gr >= 0 && gr < HH && gc >= 0 && gc < WW)
                v = base[gr * WW + gc] ? (_Float16)1.f : (_Float16)0.f;
            prow[c] = v;
        }
    }
}

// ------------------------------------------------------------- kernels -------
__global__ void k_zero(float* acc) { *acc = 0.f; }

// Per-pixel categorical sampling (stateless hash RNG, inverse CDF), S samples.
__global__ void k_sample(const float* __restrict__ preds,
                         unsigned char* __restrict__ samp) {
    const int idx = blockIdx.x * blockDim.x + threadIdx.x;
    if (idx >= NIMG * HWP) return;
    const int img = idx / HWP, pix = idx - img * HWP;
    const float* p = preds + (size_t)img * NCLSS * HWP + pix;
    const float c0 = p[0];
    const float c1 = c0 + p[(size_t)HWP];
    const float c2 = c1 + p[(size_t)2 * HWP];
    const float tot = c2 + p[(size_t)3 * HWP];
#pragma unroll
    for (int s = 0; s < NSAMP; ++s) {
        const unsigned h = pcg_hash(pcg_hash((unsigned)(idx * NSAMP + s)) ^ 0x9E3779B9u);
        const float u = (h >> 8) * (1.0f / 16777216.0f) * tot;
        const int cc = (u >= c0) + (u >= c1) + (u >= c2);
        samp[((size_t)img * NSAMP + s) * HWP + pix] = (unsigned char)cc;
    }
}

// 5x5 box-count * fg via WMMA + per-tile argmax (first-max tie-break).
__global__ void __launch_bounds__(32)
k_boxsum5_argmax(const unsigned char* __restrict__ samp,
                 float* __restrict__ bmv, int* __restrict__ bmi, int cls) {
    __shared__ __attribute__((aligned(32))) _Float16 patch[32 * 32];
    __shared__ __attribute__((aligned(32))) _Float16 rs[16 * 32];
    const int plane = blockIdx.z;
    const int row0 = blockIdx.y * 16, col0 = blockIdx.x * 16;
    const unsigned char* base = samp + (size_t)plane * HWP;
    const int lane = threadIdx.x & 31;
    load_mask_row(patch + lane * 32, base, row0 - 8 + lane, col0 - 8, cls);
    __syncthreads();
    v8f d = boxsum_wmma<2>(patch, rs);
    const int m = lane & 15, hf = lane >> 4;
    float bestv = -1.f; int besti = 0;
#pragma unroll
    for (int v = 0; v < 8; ++v) {
        const int r = row0 + v + 8 * hf, c = col0 + m;
        const float fgc = (float)patch[(v + 8 * hf + 8) * 32 + (m + 8)];
        const float nv = d[v] * fgc;                 // count_neighbor * fg
        const int idx = r * WW + c;
        if (nv > bestv || (nv == bestv && idx < besti)) { bestv = nv; besti = idx; }
    }
#pragma unroll
    for (int off = 16; off > 0; off >>= 1) {
        const float ov = __shfl_xor(bestv, off, 32);
        const int   oi = __shfl_xor(besti, off, 32);
        if (ov > bestv || (ov == bestv && oi < besti)) { bestv = ov; besti = oi; }
    }
    if (lane == 0) {
        const int t = blockIdx.y * TILES_X + blockIdx.x;
        bmv[plane * NTILES + t] = bestv;
        bmi[plane * NTILES + t] = besti;
    }
}

__global__ void k_seed_reduce(const float* __restrict__ bmv,
                              const int* __restrict__ bmi,
                              int* __restrict__ seeds, int* __restrict__ hasfg) {
    __shared__ float sv[256];
    __shared__ int   si[256];
    const int plane = blockIdx.x, tid = threadIdx.x;
    float bv = -1.f; int bi = 0;
    for (int t = tid; t < NTILES; t += 256) {
        const float v = bmv[plane * NTILES + t];
        const int   i = bmi[plane * NTILES + t];
        if (v > bv || (v == bv && i < bi)) { bv = v; bi = i; }
    }
    sv[tid] = bv; si[tid] = bi; __syncthreads();
    for (int s = 128; s > 0; s >>= 1) {
        if (tid < s) {
            if (sv[tid + s] > sv[tid] ||
                (sv[tid + s] == sv[tid] && si[tid + s] < si[tid])) {
                sv[tid] = sv[tid + s]; si[tid] = si[tid + s];
            }
        }
        __syncthreads();
    }
    if (tid == 0) { seeds[plane] = si[0]; hasfg[plane] = (sv[0] > 0.f) ? 1 : 0; }
}

__global__ void k_reach_init(unsigned int* __restrict__ reach) {
    const size_t i = (size_t)blockIdx.x * blockDim.x + threadIdx.x;
    if (i < (size_t)NPLANE * HWP / 4) reach[i] = 0u;
}

__global__ void k_seed_set(unsigned char* __restrict__ reach,
                           const int* __restrict__ seeds,
                           const int* __restrict__ hasfg) {
    const int p = threadIdx.x;
    if (p < NPLANE && hasfg[p]) reach[(size_t)p * HWP + seeds[p]] = 1;
}

// One flood-fill pass: 32x32 LDS tile with 1-pixel global halo, iterate masked
// 8-dilation to local fixed point (monotone -> in-place is safe), write back.
// The fg byte tile is DMA'd into LDS with GLOBAL_LOAD_ASYNC_TO_LDS_B128
// (GVS mode, ASYNCcnt-tracked), then converted to 0/1 in place.
__global__ void __launch_bounds__(256)
k_flood(unsigned char* __restrict__ reach,
        const unsigned char* __restrict__ samp, int cls) {
    __shared__ unsigned char t[34][36];
    __shared__ __attribute__((aligned(16))) unsigned char fg[32][32];
    __shared__ int changed;
    const int plane = blockIdx.z;
    const int row0 = blockIdx.y * 32, col0 = blockIdx.x * 32;
    unsigned char* rbase = reach + (size_t)plane * HWP;
    const unsigned char* sbase = samp + (size_t)plane * HWP;
    const int tid = threadIdx.y * 32 + threadIdx.x;

    // 64 lanes x 16B async copies cover the 32x32 fg tile (always in-bounds,
    // 16B-aligned on both sides: col0 % 32 == 0, row stride 32).
    if (tid < 64) {
        const int r = tid >> 1, c = (tid & 1) * 16;
        const unsigned voff = (unsigned)((row0 + r) * WW + (col0 + c));
        const unsigned lds = (unsigned)(unsigned long long)(const void*)&fg[r][c];
        asm volatile("global_load_async_to_lds_b128 %0, %1, %2"
                     :: "v"(lds), "v"(voff),
                        "s"((unsigned long long)(size_t)sbase)
                     : "memory");
    }
    // reach halo (clamped OOB -> 0) via normal loads, overlapping the DMA.
    for (int i = tid; i < 34 * 34; i += 256) {
        const int r = i / 34, c = i % 34;
        const int gr = row0 - 1 + r, gc = col0 - 1 + c;
        t[r][c] = (gr >= 0 && gr < HH && gc >= 0 && gc < WW) ? rbase[gr * WW + gc]
                                                             : (unsigned char)0;
    }
    asm volatile("s_wait_asynccnt 0x0" ::: "memory");
    __syncthreads();
#pragma unroll
    for (int k = 0; k < 4; ++k) {    // class byte -> 0/1, each thread owns pixels
        const int r = threadIdx.y + 8 * k, c = threadIdx.x;
        fg[r][c] = (fg[r][c] == (unsigned char)cls) ? 1 : 0;
    }
    __syncthreads();
    for (int it = 0; it < 34; ++it) {       // tile diameter bound
        if (tid == 0) changed = 0;
        __syncthreads();
#pragma unroll
        for (int k = 0; k < 4; ++k) {
            const int r = threadIdx.y + 8 * k;
            const int c = threadIdx.x;
            if (!t[r + 1][c + 1] && fg[r][c]) {
                const unsigned char nb = t[r][c] | t[r][c + 1] | t[r][c + 2]
                                       | t[r + 1][c] | t[r + 1][c + 2]
                                       | t[r + 2][c] | t[r + 2][c + 1] | t[r + 2][c + 2];
                if (nb) { t[r + 1][c + 1] = 1; changed = 1; }
            }
        }
        __syncthreads();
        const int ch = changed;
        __syncthreads();
        if (!ch) break;
    }
    for (int k = 0; k < 4; ++k) {
        const int r = threadIdx.y + 8 * k, c = threadIdx.x;
        rbase[(row0 + r) * WW + col0 + c] = t[r + 1][c + 1];  // monotone overwrite
    }
}

// 3x3 box counts of fill and fg via WMMA, binary reward, fused with the
// REINFORCE dot product r * log(prob_ns + eps) -> global accumulator.
__global__ void __launch_bounds__(32)
k_reward(const unsigned char* __restrict__ reach,
         const unsigned char* __restrict__ samp,
         const float* __restrict__ preds, float* __restrict__ acc, int cls) {
    __shared__ __attribute__((aligned(32))) _Float16 pF[32 * 32];
    __shared__ __attribute__((aligned(32))) _Float16 pG[32 * 32];
    __shared__ __attribute__((aligned(32))) _Float16 rs[16 * 32];
    const int plane = blockIdx.z;
    const int row0 = blockIdx.y * 16, col0 = blockIdx.x * 16;
    const unsigned char* rbase = reach + (size_t)plane * HWP;
    const unsigned char* sbase = samp + (size_t)plane * HWP;
    const int lane = threadIdx.x & 31;
    load_bin_row(pF + lane * 32, rbase, row0 - 8 + lane, col0 - 8);
    load_mask_row(pG + lane * 32, sbase, row0 - 8 + lane, col0 - 8, cls);
    __syncthreads();
    v8f dF = boxsum_wmma<1>(pF, rs);
    v8f dG = boxsum_wmma<1>(pG, rs);
    const int m = lane & 15, hf = lane >> 4;
    const int img = plane / NSAMP;
    float sum = 0.f;
#pragma unroll
    for (int v = 0; v < 8; ++v) {
        const int r = row0 + v + 8 * hf, c = col0 + m;
        const float fillc = (float)pF[(v + 8 * hf + 8) * 32 + m + 8];
        const float fgc   = (float)pG[(v + 8 * hf + 8) * 32 + m + 8];
        const float Fc = dF[v] * fillc;           // F_fg_neighbors (exact ints)
        const float Sc = dG[v] * fgc;             // S_fg_neighbors
        const float f2 = (Fc == 0.f) ? -1.f : Fc; // where(Fc==0,-1,Fc)
        const float s2 = (Sc == 0.f) ? -2.f : Sc; // where(Sc==0,-2,Sc)
        const float rwd = (f2 == s2) ? 1.f : 0.f;
        const int o = r * WW + c;
        const int cc = (int)sbase[o];
        const float p = preds[((size_t)img * NCLSS + cc) * HWP + o]; // prob_ns
        sum += rwd * __logf(p + 1e-16f);
    }
#pragma unroll
    for (int off = 16; off > 0; off >>= 1) sum += __shfl_xor(sum, off, 32);
    if (lane == 0) atomicAdd(acc, sum);
}

__global__ void k_final(const float* __restrict__ acc, float* __restrict__ out) {
    out[0] = -acc[0] / (float)((size_t)NIMG * NSAMP * HWP);
}

// ------------------------------------------------------------- driver --------
extern "C" void kernel_launch(void* const* d_in, const int* in_sizes, int n_in,
                              void* d_out, int out_size, void* d_ws, size_t ws_size,
                              hipStream_t stream) {
    (void)in_sizes; (void)n_in; (void)out_size; (void)ws_size;
    const float* preds = (const float*)d_in[0];
    float* out = (float*)d_out;

    // Workspace carve-up (~21.3 MB total)
    char* ws = (char*)d_ws;
    float* acc   = (float*)ws;                                   // 4 B
    int*   seeds = (int*)(ws + 256);                             // 40*4
    int*   hasfg = (int*)(ws + 512);                             // 40*4
    float* bmv   = (float*)(ws + 1024);                          // 40*1024*4
    int*   bmi   = (int*)(ws + 1024 + (size_t)NPLANE * NTILES * 4);
    unsigned char* samp  = (unsigned char*)(ws + 1024 + (size_t)NPLANE * NTILES * 8);
    unsigned char* reach = samp + (size_t)NPLANE * HWP;

    k_zero<<<1, 1, 0, stream>>>(acc);
    k_sample<<<(NIMG * HWP + 255) / 256, 256, 0, stream>>>(preds, samp);

    for (int cls = 1; cls < NCLSS; ++cls) {
        k_boxsum5_argmax<<<dim3(TILES_X, TILES_Y, NPLANE), 32, 0, stream>>>(
            samp, bmv, bmi, cls);
        k_seed_reduce<<<NPLANE, 256, 0, stream>>>(bmv, bmi, seeds, hasfg);
        k_reach_init<<<(int)(((size_t)NPLANE * HWP / 4 + 255) / 256), 256, 0, stream>>>(
            (unsigned int*)reach);
        k_seed_set<<<1, 64, 0, stream>>>(reach, seeds, hasfg);
        for (int it = 0; it < FLOOD_OUTER; ++it)
            k_flood<<<dim3(WW / 32, HH / 32, NPLANE), dim3(32, 8), 0, stream>>>(
                reach, samp, cls);
        k_reward<<<dim3(TILES_X, TILES_Y, NPLANE), 32, 0, stream>>>(
            reach, samp, preds, acc, cls);
    }
    k_final<<<1, 1, 0, stream>>>(acc, out);
}
